// ScaledDotProductAttention_33930241639016
// MI455X (gfx1250) — matrix operand order repaired and checked by reading, not compiled, not run on hardware
//
#include <hip/hip_runtime.h>

typedef _Float16 v16h __attribute__((ext_vector_type(16)));
typedef _Float16 v8h  __attribute__((ext_vector_type(8)));
typedef _Float16 v4h  __attribute__((ext_vector_type(4)));
typedef _Float16 v2h  __attribute__((ext_vector_type(2)));
typedef float    v8f  __attribute__((ext_vector_type(8)));
typedef float    v4f  __attribute__((ext_vector_type(4)));

#define D_HEAD   64
#define BLOCK_M  64
#define BLOCK_N  64
#define NWAVES   4
#define NTHREADS 128
#define NEG_INF  (-1e20f)

// Native base-2 exp: lowers to v_exp_f32.
__device__ __forceinline__ float fast_exp2(float x) {
    return __builtin_amdgcn_exp2f(x);
}

// Flash attention forward: fp32 in/out, f16 WMMA (f32 accumulate) for both GEMMs,
// base-2 online softmax in fp32, per-batch kv-length masking.
// - K and V tiles staged cooperatively in LDS as f16 (V transposed): every WMMA
//   B-fragment is two 16B ds loads.
// - 1/sqrt(d)*log2(e) folded into the one-time Q->f16 conversion.
// - Row sums of P computed on the matrix pipe (P x ones); result lands in
//   C-layout, matching the per-lane row bookkeeping.
// - KV loop stops at the last block containing a valid key: beyond it every
//   probability underflows to exactly 0 and the running max is unchanged, so
//   skipping is bit-exact. vlen==0 (reference: uniform over ALL keys) falls
//   back to the full loop, which the mask path reproduces naturally.
__global__ __launch_bounds__(NTHREADS)
void ScaledDotProductAttention_kernel(const float* __restrict__ Q,
                                      const float* __restrict__ K,
                                      const float* __restrict__ V,
                                      const int*   __restrict__ vlens,
                                      float* __restrict__ O,
                                      int NQ, int NKV)
{
    __shared__ _Float16 sK [BLOCK_N * D_HEAD];          // [krow][d]   8 KB
    __shared__ _Float16 sVt[D_HEAD * BLOCK_N];          // [vcol][kv]  8 KB
    __shared__ _Float16 sP [NWAVES * 16 * BLOCK_N];     // per-wave P  8 KB

    const int b      = blockIdx.y;
    const int qTile  = blockIdx.x;
    const int tid    = threadIdx.x;
    const int wave   = tid >> 5;
    const int lane   = tid & 31;
    const int laneLo = lane & 15;
    const int laneHi = lane >> 4;            // 0 or 1

    const int vlen  = vlens[b];
    const int qRow0 = qTile * BLOCK_M + wave * 16;

    // last block with any valid key (vlen==0 -> full range for uniform softmax)
    const int kvEnd = (vlen == 0) ? NKV
                    : min(NKV, ((vlen + BLOCK_N - 1) / BLOCK_N) * BLOCK_N);

    // 1/sqrt(64) * log2(e): S out of the WMMA is already in the exp2 domain.
    const float QSCALE = 0.125f * 1.44269504088896340736f;

    // ---------- load Q fragments once (A layout, two 16x32 f16 chunks over D=64) ----------
    const float* qptr = Q + ((size_t)b * NQ + qRow0 + laneLo) * D_HEAD;
    v16h aQ[2];
    #pragma unroll
    for (int c = 0; c < 2; ++c) {
        #pragma unroll
        for (int g = 0; g < 2; ++g) {        // halves g*8 .. g*8+7
            const int kbase = c * 32 + g * 16 + laneHi * 8;
            v4f f0 = *reinterpret_cast<const v4f*>(qptr + kbase);
            v4f f1 = *reinterpret_cast<const v4f*>(qptr + kbase + 4);
            #pragma unroll
            for (int j = 0; j < 4; ++j) {
                aQ[c][g*8 + j]     = (_Float16)(f0[j] * QSCALE);
                aQ[c][g*8 + 4 + j] = (_Float16)(f1[j] * QSCALE);
            }
        }
    }

    // all-ones B fragment: row-sum via the matrix pipe
    v16h bOnes;
    #pragma unroll
    for (int j = 0; j < 16; ++j) bOnes[j] = (_Float16)1.0f;

    float m[8], l[8];
    v8f   accO[4];
    #pragma unroll
    for (int r = 0; r < 8; ++r) { m[r] = -INFINITY; l[r] = 0.0f; }
    #pragma unroll
    for (int t = 0; t < 4; ++t) accO[t] = {};

    _Float16* sPw = &sP[wave * 16 * BLOCK_N];

    for (int kv0 = 0; kv0 < kvEnd; kv0 += BLOCK_N) {
        __syncthreads();   // previous iteration's readers done with sK / sVt

        const float* kbase_g = K + ((size_t)b * NKV + kv0) * D_HEAD;
        const float* vbase_g = V + ((size_t)b * NKV + kv0) * D_HEAD;

        // ---------- stage K tile (f32 -> f16, natural layout, packed b64 stores) ----------
        #pragma unroll
        for (int i = 0; i < 8; ++i) {
            const int idx4 = tid + i * NTHREADS;    // float4 index in 64x64 tile
            const int row  = idx4 >> 4;
            const int col4 = idx4 & 15;
            v4f v = *reinterpret_cast<const v4f*>(kbase_g + row * D_HEAD + col4 * 4);
            v4h h;
            h[0] = (_Float16)v[0]; h[1] = (_Float16)v[1];
            h[2] = (_Float16)v[2]; h[3] = (_Float16)v[3];
            *reinterpret_cast<v4h*>(&sK[row * D_HEAD + col4 * 4]) = h;
        }

        // ---------- stage V tile transposed (pack pairs along kv, b32 stores) ----------
        #pragma unroll
        for (int i = 0; i < 4; ++i) {
            const int idx  = tid + i * NTHREADS;    // 0..511 work items
            const int rp   = idx >> 4;              // kv row pair 0..31
            const int col4 = idx & 15;
            const float* p0 = vbase_g + (2 * rp) * D_HEAD + col4 * 4;
            v4f a0 = *reinterpret_cast<const v4f*>(p0);
            v4f a1 = *reinterpret_cast<const v4f*>(p0 + D_HEAD);
            #pragma unroll
            for (int j = 0; j < 4; ++j) {
                v2h h;
                h[0] = (_Float16)a0[j];             // kv = 2*rp
                h[1] = (_Float16)a1[j];             // kv = 2*rp + 1
                *reinterpret_cast<v2h*>(&sVt[(col4*4 + j) * BLOCK_N + 2 * rp]) = h;
            }
        }

        // prefetch next tiles (one cacheline per thread each)
        if (kv0 + BLOCK_N < kvEnd) {
            __builtin_prefetch(kbase_g + BLOCK_N * D_HEAD + tid * 32, 0, 0);
            __builtin_prefetch(vbase_g + BLOCK_N * D_HEAD + tid * 32, 0, 0);
        }
        __syncthreads();

        // ---------- S = (Q K^T) * scale for 4 column tiles of 16 ----------
        v8f s[4];
        #pragma unroll
        for (int t = 0; t < 4; ++t) {
            v8f acc = {};
            #pragma unroll
            for (int c = 0; c < 2; ++c) {
                const _Float16* kb = &sK[(t * 16 + laneLo) * D_HEAD + c * 32 + laneHi * 8];
                const v16h bK = __builtin_shufflevector(*reinterpret_cast<const v8h*>(kb),
                    *reinterpret_cast<const v8h*>(kb + 16), 0,1,2,3,4,5,6,7,8,9,10,11,12,13,14,15);
                acc = __builtin_amdgcn_wmma_f32_16x16x32_f16(false, aQ[c], false, bK,
                                                             (short)0, acc, false, false);
            }
            s[t] = acc;
        }

        // kv-length mask (column-dependent only)
        #pragma unroll
        for (int t = 0; t < 4; ++t) {
            const int col = kv0 + t * 16 + laneLo;
            const bool ok = col < vlen;
            #pragma unroll
            for (int r = 0; r < 8; ++r)
                s[t][r] = ok ? s[t][r] : NEG_INF;
        }

        // ---------- online softmax (base 2), emit P (f16) into per-wave LDS ----------
        #pragma unroll
        for (int r = 0; r < 8; ++r) {
            float x = fmaxf(fmaxf(s[0][r], s[1][r]), fmaxf(s[2][r], s[3][r]));
            #pragma unroll
            for (int off = 8; off >= 1; off >>= 1)
                x = fmaxf(x, __shfl_xor(x, off, 16));   // row lives in a 16-lane group
            const float mnew  = fmaxf(m[r], x);
            const float alpha = fast_exp2(m[r] - mnew);
            m[r]  = mnew;
            l[r] *= alpha;
            const float p0 = fast_exp2(s[0][r] - mnew);
            const float p1 = fast_exp2(s[1][r] - mnew);
            const float p2 = fast_exp2(s[2][r] - mnew);
            const float p3 = fast_exp2(s[3][r] - mnew);
            #pragma unroll
            for (int t = 0; t < 4; ++t) accO[t][r] *= alpha;

            const int prow = r + laneHi * 8;            // C-layout row -> LDS row
            sPw[prow * BLOCK_N +  0 + laneLo] = (_Float16)p0;
            sPw[prow * BLOCK_N + 16 + laneLo] = (_Float16)p1;
            sPw[prow * BLOCK_N + 32 + laneLo] = (_Float16)p2;
            sPw[prow * BLOCK_N + 48 + laneLo] = (_Float16)p3;
        }

        // wave-private LDS region: stores precede loads, LDS in-order within wave
        asm volatile("s_wait_dscnt 0" ::: "memory");

        // ---------- reload P in A layout ----------
        v16h aP[2];
        #pragma unroll
        for (int c = 0; c < 2; ++c) {
            const _Float16* base = &sPw[laneLo * BLOCK_N + c * 32 + laneHi * 8];
            v8h lo = *reinterpret_cast<const v8h*>(base);
            v8h hi = *reinterpret_cast<const v8h*>(base + 16);
            aP[c] = __builtin_shufflevector(lo, hi,
                     0,1,2,3,4,5,6,7,8,9,10,11,12,13,14,15);
        }

        // ---------- row sums on the matrix pipe: l += P x ones ----------
        v8f accL = {};
        accL = __builtin_amdgcn_wmma_f32_16x16x32_f16(false, aP[0], false, bOnes,
                                                      (short)0, accL, false, false);
        accL = __builtin_amdgcn_wmma_f32_16x16x32_f16(false, aP[1], false, bOnes,
                                                      (short)0, accL, false, false);
        #pragma unroll
        for (int r = 0; r < 8; ++r) l[r] += accL[r];

        // ---------- O += P * V  (V B-fragments straight out of transposed LDS) ----------
        #pragma unroll
        for (int t = 0; t < 4; ++t) {
            #pragma unroll
            for (int c = 0; c < 2; ++c) {
                const _Float16* vb = &sVt[(t * 16 + laneLo) * BLOCK_N + c * 32 + laneHi * 8];
                const v16h bV = __builtin_shufflevector(*reinterpret_cast<const v8h*>(vb),
                    *reinterpret_cast<const v8h*>(vb + 16), 0,1,2,3,4,5,6,7,8,9,10,11,12,13,14,15);
                accO[t] = __builtin_amdgcn_wmma_f32_16x16x32_f16(false, aP[c], false, bV,
                                                                 (short)0, accO[t], false, false);
            }
        }
    }

    // ---------- epilogue: divide by l, store fp32 ----------
    #pragma unroll
    for (int r = 0; r < 8; ++r) {
        const float inv = 1.0f / l[r];
        const int row = qRow0 + r + laneHi * 8;
        float* optr = O + ((size_t)b * NQ + row) * D_HEAD;
        #pragma unroll
        for (int t = 0; t < 4; ++t)
            optr[t * 16 + laneLo] = accO[t][r] * inv;
    }
}

extern "C" void kernel_launch(void* const* d_in, const int* in_sizes, int n_in,
                              void* d_out, int out_size, void* d_ws, size_t ws_size,
                              hipStream_t stream) {
    const float* Q  = (const float*)d_in[0];
    const float* K  = (const float*)d_in[1];
    const float* V  = (const float*)d_in[2];
    const int* vlen = (const int*)d_in[3];
    float* O        = (float*)d_out;

    const int B   = in_sizes[3];
    const int NQ  = in_sizes[0] / (B * D_HEAD);
    const int NKV = in_sizes[1] / (B * D_HEAD);

    dim3 grid(NQ / BLOCK_M, B);
    dim3 block(NTHREADS);
    ScaledDotProductAttention_kernel<<<grid, block, 0, stream>>>(Q, K, V, vlen, O, NQ, NKV);
}
